// ByteMoELayer_25512105739054
// MI455X (gfx1250) — matrix operand order repaired and checked by reading, not compiled, not run on hardware
//
#include <hip/hip_runtime.h>
#include <stdint.h>

typedef unsigned short ushort_t;
typedef __attribute__((ext_vector_type(16))) __bf16 v16bf;
typedef __attribute__((ext_vector_type(8)))  float  v8f;

#define B_   4
#define S_   2048
#define H_   2048
#define E_   8
#define F_   5632
#define C_   64
#define N_   8192      // B_*S_
#define K_   2
#define CAP_ 2560      // int(1.25 * N * K / E)
#define LDT  40        // LDS tile row stride in ushorts (32 + 8 pad, keeps 16B alignment)

__device__ __forceinline__ ushort_t f2bf(float f) {
  unsigned int u = __float_as_uint(f);
  u += 0x7FFFu + ((u >> 16) & 1u);          // round-to-nearest-even
  return (ushort_t)(u >> 16);
}

__device__ __forceinline__ unsigned int rne_hi(float f) {
  unsigned int u = __float_as_uint(f);
  return u + (0x7FFFu + ((u >> 16) & 1u));  // rounded bf16 sits in bits [31:16]
}

// pack two floats to packed bf16 {hi,lo}: one v_perm_b32 merges the halves
__device__ __forceinline__ unsigned int pk2bf(float lo, float hi) {
  return __builtin_amdgcn_perm(rne_hi(hi), rne_hi(lo), 0x07060302u);
}

union Frag16 { uint4 q[2]; v16bf v; };
union Pack8  { ushort_t s[8]; uint4 q; };

// ---------------------------------------------------------------------------
// K0: ctx_logits[b][e] = tanh(prev_hidden @ Wctx) @ Wctx2     (tiny)
// ---------------------------------------------------------------------------
__global__ __launch_bounds__(256) void ctx_kernel(
    const float* __restrict__ prev, const float* __restrict__ Wctx,
    const float* __restrict__ Wctx2, float* __restrict__ ctx_logits) {
  __shared__ float ctx[B_][C_];
  int tid = threadIdx.x;
  int b = tid >> 6, c = tid & 63;
  float acc = 0.f;
  for (int i = 0; i < H_; ++i) acc += prev[b * H_ + i] * Wctx[i * C_ + c];
  ctx[b][c] = tanhf(acc);
  __syncthreads();
  if (tid < B_ * E_) {
    int bb = tid >> 3, e = tid & 7;
    float a = 0.f;
    for (int cc = 0; cc < C_; ++cc) a += ctx[bb][cc] * Wctx2[cc * E_ + e];
    ctx_logits[bb * E_ + e] = a;
  }
}

// ---------------------------------------------------------------------------
// K1: per-token rmsnorm + router logits + softmax + top2 + gates (1 block/token)
// ---------------------------------------------------------------------------
__global__ __launch_bounds__(256) void router_kernel(
    const float* __restrict__ x, const float* __restrict__ rms_w,
    const float* __restrict__ Wg, const float* __restrict__ ctx_logits,
    float* __restrict__ probs, int* __restrict__ topi, float* __restrict__ gate) {
  int n = blockIdx.x;
  int tid = threadIdx.x;
  const float* xr = x + (size_t)n * H_;

  float ss = 0.f;
  float acc[E_];
#pragma unroll
  for (int e = 0; e < E_; ++e) acc[e] = 0.f;

#pragma unroll
  for (int i = 0; i < 8; ++i) {
    int gi = tid * 8 + i;
    float xv = xr[gi];
    float wv = rms_w[gi];
    ss += xv * xv;
    float xw = xv * wv;                       // rms scalar factored out
    const float* wgrow = Wg + (size_t)gi * E_;
#pragma unroll
    for (int e = 0; e < E_; ++e) acc[e] += xw * wgrow[e];
  }

  __shared__ float red[256];
  __shared__ float res[9];
#pragma unroll
  for (int q = 0; q < 9; ++q) {
    red[tid] = (q == 0) ? ss : acc[q - 1];
    __syncthreads();
    for (int s = 128; s > 0; s >>= 1) {
      if (tid < s) red[tid] += red[tid + s];
      __syncthreads();
    }
    if (tid == 0) res[q] = red[0];
    __syncthreads();
  }

  if (tid == 0) {
    float rms = rsqrtf(res[0] / (float)H_ + 1e-6f);
    int b = n / S_;
    float lg[E_];
    float mx = -1e30f;
#pragma unroll
    for (int e = 0; e < E_; ++e) {
      lg[e] = rms * res[1 + e] + ctx_logits[b * E_ + e];
      mx = fmaxf(mx, lg[e]);
    }
    float sum = 0.f;
#pragma unroll
    for (int e = 0; e < E_; ++e) { lg[e] = __expf(lg[e] - mx); sum += lg[e]; }
    float inv = 1.f / sum;
    float bv = -1.f, sv = -1.f; int bi = 0, si = 0;
#pragma unroll
    for (int e = 0; e < E_; ++e) {
      float p = lg[e] * inv;
      probs[(size_t)n * E_ + e] = p;
      if (p > bv)      { sv = bv; si = bi; bv = p; bi = e; }
      else if (p > sv) { sv = p; si = e; }
    }
    float g = 1.f / (bv + sv);
    topi[n * 2] = bi;  topi[n * 2 + 1] = si;
    gate[n * 2] = bv * g; gate[n * 2 + 1] = sv * g;
  }
}

// ---------------------------------------------------------------------------
// K2: deterministic k-major cumulative slot assignment (reference semantics)
// ---------------------------------------------------------------------------
__global__ __launch_bounds__(256) void slot_kernel(
    const int* __restrict__ topi, int* __restrict__ bufpos, int* __restrict__ counts) {
  __shared__ int se[256];
  int tid = threadIdx.x;
  int mycnt = 0;
  for (int it = 0; it < (N_ * K_) / 256; ++it) {
    int j = it * 256 + tid;                 // j = k*N + n (k-major)
    int k = j >> 13, n = j & (N_ - 1);
    se[tid] = topi[n * 2 + k];
    __syncthreads();
    if (tid < E_) {
      for (int i = 0; i < 256; ++i)
        if (se[i] == tid) bufpos[it * 256 + i] = mycnt++;
    }
    __syncthreads();
  }
  if (tid < E_) counts[tid] = mycnt;
}

// ---------------------------------------------------------------------------
// K4: dispatch gate*x (raw x) into bf16 buffers [E][CAP][H]
// ---------------------------------------------------------------------------
__global__ __launch_bounds__(256) void dispatch_kernel(
    const float* __restrict__ x, const int* __restrict__ topi,
    const float* __restrict__ gate, const int* __restrict__ bufpos,
    ushort_t* __restrict__ buffers) {
  int j = blockIdx.x;
  int pos = bufpos[j];
  if (pos >= CAP_) return;
  int k = j >> 13, n = j & (N_ - 1);
  int e = topi[n * 2 + k];
  float g = gate[n * 2 + k];
  int tid = threadIdx.x;
  const float4* xr = (const float4*)(x + (size_t)n * H_);
  ushort_t* dst = buffers + ((size_t)e * CAP_ + pos) * H_;
  float4 v0 = xr[tid * 2], v1 = xr[tid * 2 + 1];
  Pack8 p;
  p.s[0] = f2bf(g * v0.x); p.s[1] = f2bf(g * v0.y);
  p.s[2] = f2bf(g * v0.z); p.s[3] = f2bf(g * v0.w);
  p.s[4] = f2bf(g * v1.x); p.s[5] = f2bf(g * v1.y);
  p.s[6] = f2bf(g * v1.z); p.s[7] = f2bf(g * v1.w);
  ((uint4*)dst)[tid] = p.q;
}

// ---------------------------------------------------------------------------
// K5: GEMM1 fused  h = silu(A@w1) * (A@w3),  A bf16 [CAP][H], w fp32 [H][F]
// block tile 128x128, 8 waves (2Mx4N), wave tile 64x32 (both matrices share A
// fragments): 16 WMMA per K-step per wave.  Double-buffered LDS, packed b32
// weight staging, one barrier per K-step.
// ---------------------------------------------------------------------------
__global__ __launch_bounds__(256) void gemm1_kernel(
    const ushort_t* __restrict__ buffers, const float* __restrict__ w1,
    const float* __restrict__ w3, ushort_t* __restrict__ hbuf,
    const int* __restrict__ counts) {
  int e = blockIdx.z;
  int used = counts[e]; if (used > CAP_) used = CAP_;
  int m0 = blockIdx.y * 128;
  if (m0 >= used) return;                     // skip empty capacity rows
  int n0 = blockIdx.x * 128;

  const ushort_t* A  = buffers + (size_t)e * CAP_ * H_;
  const float*    B1 = w1 + (size_t)e * H_ * F_;
  const float*    B3 = w3 + (size_t)e * H_ * F_;
  ushort_t*       OutH = hbuf + (size_t)e * CAP_ * F_;

  __shared__ ushort_t As[2][128 * LDT];
  __shared__ ushort_t B1s[2][128 * LDT];
  __shared__ ushort_t B3s[2][128 * LDT];

  int tid = threadIdx.x;
  int lane = tid & 31, wave = tid >> 5;
  int lrow = lane & 15, lhalf = lane >> 4;
  int wm = wave & 1, wn = wave >> 1;          // 2 M-groups of 64, 4 N-groups of 32

  // loaders: A tile 128x32 bf16, 2x16B per thread
  int arow = tid >> 1, ahalf = tid & 1;
  // loaders: B tiles 32x128 fp32 -> each thread: 2 K-rows x 8 N-cols, packed b32
  int kp = tid >> 4, ng = tid & 15;

  const ushort_t* gA  = A + (size_t)(m0 + arow) * H_ + ahalf * 16;
  const float*    gB1 = B1 + (size_t)(2 * kp) * F_ + n0 + ng * 8;
  const float*    gB3 = B3 + (size_t)(2 * kp) * F_ + n0 + ng * 8;

  const int KT = H_ / 32;

  v8f zero = {};
  v8f acc1[4][2], acc3[4][2];
#pragma unroll
  for (int mt = 0; mt < 4; ++mt)
#pragma unroll
    for (int nt = 0; nt < 2; ++nt) { acc1[mt][nt] = zero; acc3[mt][nt] = zero; }

  uint4 ra0, ra1;
  float4 P[4], Q[4];   // P: w1 rows {2kp,2kp+1} x 8 cols; Q: w3 likewise

  auto load_tile = [&](int kt) {
    const uint4* pa = (const uint4*)(gA + (size_t)kt * 32);
    ra0 = pa[0]; ra1 = pa[1];
    const float* s1 = gB1 + (size_t)kt * 32 * F_;
    P[0] = *(const float4*)(s1);
    P[1] = *(const float4*)(s1 + 4);
    P[2] = *(const float4*)(s1 + F_);
    P[3] = *(const float4*)(s1 + F_ + 4);
    const float* s3 = gB3 + (size_t)kt * 32 * F_;
    Q[0] = *(const float4*)(s3);
    Q[1] = *(const float4*)(s3 + 4);
    Q[2] = *(const float4*)(s3 + F_);
    Q[3] = *(const float4*)(s3 + F_ + 4);
  };
  auto store_tile = [&](int buf) {
    uint4* sa = (uint4*)(&As[buf][arow * LDT + ahalf * 16]);
    sa[0] = ra0; sa[1] = ra1;
    float r0[8] = {P[0].x, P[0].y, P[0].z, P[0].w, P[1].x, P[1].y, P[1].z, P[1].w};
    float r1[8] = {P[2].x, P[2].y, P[2].z, P[2].w, P[3].x, P[3].y, P[3].z, P[3].w};
    float s0[8] = {Q[0].x, Q[0].y, Q[0].z, Q[0].w, Q[1].x, Q[1].y, Q[1].z, Q[1].w};
    float s1[8] = {Q[2].x, Q[2].y, Q[2].z, Q[2].w, Q[3].x, Q[3].y, Q[3].z, Q[3].w};
#pragma unroll
    for (int i = 0; i < 8; ++i) {               // [N][K] layout, K packed in pairs
      *(unsigned int*)&B1s[buf][(ng * 8 + i) * LDT + 2 * kp] = pk2bf(r0[i], r1[i]);
      *(unsigned int*)&B3s[buf][(ng * 8 + i) * LDT + 2 * kp] = pk2bf(s0[i], s1[i]);
    }
  };

  load_tile(0);
  store_tile(0);
  __syncthreads();

  for (int kt = 0; kt < KT; ++kt) {
    int cur = kt & 1;
    bool more = (kt + 1 < KT);
    if (more) load_tile(kt + 1);                // global loads in flight over WMMAs
    if (kt + 2 < KT) {                          // gfx1250 global_prefetch_b8
      __builtin_prefetch(gA + (size_t)(kt + 2) * 32, 0, 3);
      __builtin_prefetch(gB1 + (size_t)(kt + 2) * 32 * F_, 0, 3);
      __builtin_prefetch(gB3 + (size_t)(kt + 2) * 32 * F_, 0, 3);
    }

    Frag16 af[4], bf1[2], bf3[2];
#pragma unroll
    for (int mt = 0; mt < 4; ++mt) {
      const ushort_t* base = &As[cur][(wm * 64 + mt * 16 + lrow) * LDT];
      af[mt].q[0] = *(const uint4*)(base + lhalf * 8);        // K = L*8 .. +8
      af[mt].q[1] = *(const uint4*)(base + 16 + lhalf * 8);   // K = 16+L*8 ..
    }
#pragma unroll
    for (int nt = 0; nt < 2; ++nt) {
      const ushort_t* pp1 = &B1s[cur][(wn * 32 + nt * 16 + lrow) * LDT + lhalf * 8];
      bf1[nt].q[0] = *(const uint4*)(pp1);
      bf1[nt].q[1] = *(const uint4*)(pp1 + 16);
      const ushort_t* pp3 = &B3s[cur][(wn * 32 + nt * 16 + lrow) * LDT + lhalf * 8];
      bf3[nt].q[0] = *(const uint4*)(pp3);
      bf3[nt].q[1] = *(const uint4*)(pp3 + 16);
    }

#pragma unroll
    for (int mt = 0; mt < 4; ++mt)
#pragma unroll
      for (int nt = 0; nt < 2; ++nt) {
        acc1[mt][nt] = __builtin_amdgcn_wmma_f32_16x16x32_bf16(
            false, af[mt].v, false, bf1[nt].v, (short)0, acc1[mt][nt], false, false);
        acc3[mt][nt] = __builtin_amdgcn_wmma_f32_16x16x32_bf16(
            false, af[mt].v, false, bf3[nt].v, (short)0, acc3[mt][nt], false, false);
      }

    if (more) store_tile(cur ^ 1);
    __syncthreads();
  }

  // ---- epilogue: silu(acc1) * acc3 -> bf16 h ----
#pragma unroll
  for (int mt = 0; mt < 4; ++mt)
#pragma unroll
    for (int nt = 0; nt < 2; ++nt) {
      int row = m0 + wm * 64 + mt * 16 + lhalf * 8;
      int col = n0 + wn * 32 + nt * 16 + lrow;
#pragma unroll
      for (int r = 0; r < 8; ++r) {
        float g1 = acc1[mt][nt][r];
        float g3 = acc3[mt][nt][r];
        float sg = g1 / (1.f + __expf(-g1));
        OutH[(size_t)(row + r) * F_ + col] = f2bf(sg * g3);
      }
    }
}

// ---------------------------------------------------------------------------
// K6: GEMM2  eout = h @ w2,  h bf16 [CAP][F], w2 fp32 [F][H], eout fp32
// block tile 128x256, 8 waves (2Mx4N), wave tile 64x64: 16 WMMA per K-step
// ---------------------------------------------------------------------------
__global__ __launch_bounds__(256) void gemm2_kernel(
    const ushort_t* __restrict__ hbuf, const float* __restrict__ w2,
    float* __restrict__ eout, const int* __restrict__ counts) {
  int e = blockIdx.z;
  int used = counts[e]; if (used > CAP_) used = CAP_;
  int m0 = blockIdx.y * 128;
  if (m0 >= used) return;
  int n0 = blockIdx.x * 256;

  const ushort_t* A = hbuf + (size_t)e * CAP_ * F_;
  const float*    B = w2 + (size_t)e * F_ * H_;
  float*          D = eout + (size_t)e * CAP_ * H_;

  __shared__ ushort_t As[2][128 * LDT];
  __shared__ ushort_t Bs[2][256 * LDT];

  int tid = threadIdx.x;
  int lane = tid & 31, wave = tid >> 5;
  int lrow = lane & 15, lhalf = lane >> 4;
  int wm = wave & 1, wn = wave >> 1;           // 2 M-groups of 64, 4 N-groups of 64

  int arow = tid >> 1, ahalf = tid & 1;        // A: 128x32 bf16
  int kp = tid >> 4, ng = tid & 15;            // B: 2 K-rows x 16 N-cols / thread

  const ushort_t* gA = A + (size_t)(m0 + arow) * F_ + ahalf * 16;
  const float*    gB = B + (size_t)(2 * kp) * H_ + n0 + ng * 16;

  const int KT = F_ / 32;

  v8f zero = {};
  v8f acc[4][4];
#pragma unroll
  for (int mt = 0; mt < 4; ++mt)
#pragma unroll
    for (int nt = 0; nt < 4; ++nt) acc[mt][nt] = zero;

  uint4 ra0, ra1;
  float4 R[8];                                  // rows {2kp,2kp+1} x 16 cols

  auto load_tile = [&](int kt) {
    const uint4* pa = (const uint4*)(gA + (size_t)kt * 32);
    ra0 = pa[0]; ra1 = pa[1];
    const float* s = gB + (size_t)kt * 32 * H_;
    R[0] = *(const float4*)(s);
    R[1] = *(const float4*)(s + 4);
    R[2] = *(const float4*)(s + 8);
    R[3] = *(const float4*)(s + 12);
    R[4] = *(const float4*)(s + H_);
    R[5] = *(const float4*)(s + H_ + 4);
    R[6] = *(const float4*)(s + H_ + 8);
    R[7] = *(const float4*)(s + H_ + 12);
  };
  auto store_tile = [&](int buf) {
    uint4* sa = (uint4*)(&As[buf][arow * LDT + ahalf * 16]);
    sa[0] = ra0; sa[1] = ra1;
    float r0[16] = {R[0].x, R[0].y, R[0].z, R[0].w, R[1].x, R[1].y, R[1].z, R[1].w,
                    R[2].x, R[2].y, R[2].z, R[2].w, R[3].x, R[3].y, R[3].z, R[3].w};
    float r1[16] = {R[4].x, R[4].y, R[4].z, R[4].w, R[5].x, R[5].y, R[5].z, R[5].w,
                    R[6].x, R[6].y, R[6].z, R[6].w, R[7].x, R[7].y, R[7].z, R[7].w};
#pragma unroll
    for (int i = 0; i < 16; ++i)
      *(unsigned int*)&Bs[buf][(ng * 16 + i) * LDT + 2 * kp] = pk2bf(r0[i], r1[i]);
  };

  load_tile(0);
  store_tile(0);
  __syncthreads();

  for (int kt = 0; kt < KT; ++kt) {
    int cur = kt & 1;
    bool more = (kt + 1 < KT);
    if (more) load_tile(kt + 1);
    if (kt + 2 < KT) {
      __builtin_prefetch(gA + (size_t)(kt + 2) * 32, 0, 3);
      __builtin_prefetch(gB + (size_t)(kt + 2) * 32 * H_, 0, 3);
    }

    Frag16 af[4], bfr[4];
#pragma unroll
    for (int mt = 0; mt < 4; ++mt) {
      const ushort_t* base = &As[cur][(wm * 64 + mt * 16 + lrow) * LDT];
      af[mt].q[0] = *(const uint4*)(base + lhalf * 8);
      af[mt].q[1] = *(const uint4*)(base + 16 + lhalf * 8);
    }
#pragma unroll
    for (int nt = 0; nt < 4; ++nt) {
      const ushort_t* p = &Bs[cur][(wn * 64 + nt * 16 + lrow) * LDT + lhalf * 8];
      bfr[nt].q[0] = *(const uint4*)(p);
      bfr[nt].q[1] = *(const uint4*)(p + 16);
    }

#pragma unroll
    for (int mt = 0; mt < 4; ++mt)
#pragma unroll
      for (int nt = 0; nt < 4; ++nt)
        acc[mt][nt] = __builtin_amdgcn_wmma_f32_16x16x32_bf16(
            false, af[mt].v, false, bfr[nt].v, (short)0, acc[mt][nt], false, false);

    if (more) store_tile(cur ^ 1);
    __syncthreads();
  }

#pragma unroll
  for (int mt = 0; mt < 4; ++mt)
#pragma unroll
    for (int nt = 0; nt < 4; ++nt) {
      int row = m0 + wm * 64 + mt * 16 + lhalf * 8;
      int col = n0 + wn * 64 + nt * 16 + lrow;
#pragma unroll
      for (int r = 0; r < 8; ++r)
        D[(size_t)(row + r) * H_ + col] = acc[mt][nt][r];
    }
}

// ---------------------------------------------------------------------------
// K7: combine — gather each token's K expert outputs and sum
// ---------------------------------------------------------------------------
__global__ __launch_bounds__(256) void combine_kernel(
    const float* __restrict__ eout, const int* __restrict__ topi,
    const int* __restrict__ bufpos, float* __restrict__ out) {
  int n = blockIdx.x, tid = threadIdx.x;
  float r[8];
#pragma unroll
  for (int i = 0; i < 8; ++i) r[i] = 0.f;
#pragma unroll
  for (int k = 0; k < K_; ++k) {
    int pos = bufpos[k * N_ + n];
    if (pos < CAP_) {
      int e = topi[n * 2 + k];
      const float4* src = (const float4*)(eout + ((size_t)e * CAP_ + pos) * H_) + tid * 2;
      float4 v0 = src[0], v1 = src[1];
      r[0] += v0.x; r[1] += v0.y; r[2] += v0.z; r[3] += v0.w;
      r[4] += v1.x; r[5] += v1.y; r[6] += v1.z; r[7] += v1.w;
    }
  }
  float4* dst = (float4*)(out + (size_t)n * H_) + tid * 2;
  dst[0] = make_float4(r[0], r[1], r[2], r[3]);
  dst[1] = make_float4(r[4], r[5], r[6], r[7]);
}

// ---------------------------------------------------------------------------
// K8: aux loss = 0.01 * E * sum(me * ce)
// ---------------------------------------------------------------------------
__global__ __launch_bounds__(256) void aux_kernel(
    const float* __restrict__ probs, const int* __restrict__ counts,
    float* __restrict__ out_aux) {
  int tid = threadIdx.x;
  float acc[E_];
#pragma unroll
  for (int e = 0; e < E_; ++e) acc[e] = 0.f;
  for (int n = tid; n < N_; n += 256) {
#pragma unroll
    for (int e = 0; e < E_; ++e) acc[e] += probs[(size_t)n * E_ + e];
  }
  __shared__ float red[256];
  __shared__ float me[E_];
#pragma unroll
  for (int e = 0; e < E_; ++e) {
    red[tid] = acc[e];
    __syncthreads();
    for (int s = 128; s > 0; s >>= 1) {
      if (tid < s) red[tid] += red[tid + s];
      __syncthreads();
    }
    if (tid == 0) me[e] = red[0] / (float)N_;
    __syncthreads();
  }
  if (tid == 0) {
    float aux = 0.f;
#pragma unroll
    for (int e = 0; e < E_; ++e)
      aux += me[e] * ((float)counts[e] / (float)(N_ * K_));
    out_aux[0] = 0.01f * (float)E_ * aux;
  }
}

// ---------------------------------------------------------------------------
extern "C" void kernel_launch(void* const* d_in, const int* in_sizes, int n_in,
                              void* d_out, int out_size, void* d_ws, size_t ws_size,
                              hipStream_t stream) {
  (void)in_sizes; (void)n_in; (void)out_size; (void)ws_size;
  const float* x     = (const float*)d_in[0];
  // d_in[1] = positions (unused by reference math)
  const float* prev  = (const float*)d_in[2];
  const float* rms_w = (const float*)d_in[3];
  const float* Wg    = (const float*)d_in[4];
  const float* Wctx  = (const float*)d_in[5];
  const float* Wctx2 = (const float*)d_in[6];
  const float* w1    = (const float*)d_in[7];
  const float* w2    = (const float*)d_in[8];
  const float* w3    = (const float*)d_in[9];
  float* out = (float*)d_out;

  char* ws = (char*)d_ws;
  size_t off = 0;
  auto take = [&](size_t bytes) {
    size_t o = off;
    off += (bytes + 255) & ~(size_t)255;
    return o;
  };
  float*    ctxl    = (float*)(ws + take((size_t)B_ * E_ * sizeof(float)));
  float*    probs   = (float*)(ws + take((size_t)N_ * E_ * sizeof(float)));
  int*      topi    = (int*)(ws + take((size_t)N_ * K_ * sizeof(int)));
  float*    gate    = (float*)(ws + take((size_t)N_ * K_ * sizeof(float)));
  int*      bufpos  = (int*)(ws + take((size_t)N_ * K_ * sizeof(int)));
  int*      counts  = (int*)(ws + take((size_t)E_ * sizeof(int)));
  ushort_t* buffers = (ushort_t*)(ws + take((size_t)E_ * CAP_ * H_ * sizeof(ushort_t)));
  ushort_t* hbuf    = (ushort_t*)(ws + take((size_t)E_ * CAP_ * F_ * sizeof(ushort_t)));
  float*    eout    = (float*)(ws + take((size_t)E_ * CAP_ * H_ * sizeof(float)));

  hipMemsetAsync(buffers, 0, (size_t)E_ * CAP_ * H_ * sizeof(ushort_t), stream);

  ctx_kernel<<<1, 256, 0, stream>>>(prev, Wctx, Wctx2, ctxl);
  router_kernel<<<N_, 256, 0, stream>>>(x, rms_w, Wg, ctxl, probs, topi, gate);
  slot_kernel<<<1, 256, 0, stream>>>(topi, bufpos, counts);
  dispatch_kernel<<<N_ * K_, 256, 0, stream>>>(x, topi, gate, bufpos, buffers);
  gemm1_kernel<<<dim3(F_ / 128, CAP_ / 128, E_), 256, 0, stream>>>(buffers, w1, w3, hbuf, counts);
  gemm2_kernel<<<dim3(H_ / 256, CAP_ / 128, E_), 256, 0, stream>>>(hbuf, w2, eout, counts);
  combine_kernel<<<N_, 256, 0, stream>>>(eout, topi, bufpos, out);
  aux_kernel<<<1, 256, 0, stream>>>(probs, counts, out + (size_t)N_ * H_);
}
